// SEM_SEG_Model_11261404250832
// MI455X (gfx1250) — matrix operand order repaired and checked by reading, not compiled, not run on hardware
//
#include <hip/hip_runtime.h>
#include <hip/hip_bf16.h>

// ---------------------------------------------------------------------------
// PointNet++ sem-seg pipeline for MI455X (gfx1250, wave32, WMMA).
// Heavy math (FP3 MLP, fused FP4+head MLP, ~36 GFLOP total) runs on
// v_wmma_f32_16x16x32_f16: one wave per 16-point tile, activations staged in
// LDS as f16, weights pre-packed into B-fragment lane order so each lane's
// operand is 8 contiguous dwords (L2-resident, reused by 16K tiles).
// FPS / ball-query / tiny SA MLPs are scalar VALU kernels.
// ---------------------------------------------------------------------------

#define B_   32
#define N_   8192
#define NP1  288
#define NP2  14
#define NS1  8
#define NS2  4
#define R1SQ 0.04f
#define R2SQ 0.16f
#define NCLS 13

typedef __attribute__((ext_vector_type(16))) _Float16 v16h;
typedef __attribute__((ext_vector_type(8)))  float    v8f;

union Frag {
    v16h     v;
    _Float16 h[16];
    unsigned u[8];
};

__device__ __forceinline__ float leakyf(float v) { return v > 0.0f ? v : 0.2f * v; }

// One 16x16 D tile = A(16 x 32*nkc from LDS, row-major f16) x B(packed frags).
// A-fragment lane map (contiguous K order, same as packed B): lane L: m=L&15, g=L>>4,
//   dword v holds K pair starting at 2v + 16g.
// B packed per chunk: lane L owns 16 halves = K 16g+h for column n=L&15.
__device__ __forceinline__ v8f wave_gemm_tile(const _Float16* actLDS, int lda,
                                              const _Float16* packB, int chunk0, int nkc,
                                              float bias_val, int lane)
{
    const int m16 = lane & 15;
    const int g   = lane >> 4;
    v8f acc;
#pragma unroll
    for (int v = 0; v < 8; ++v) acc[v] = bias_val;
#pragma unroll
    for (int kc = 0; kc < nkc; ++kc) {
        Frag a, bf;
#pragma unroll
        for (int v = 0; v < 8; ++v) {
            int k = kc * 32 + 2 * v + 16 * g;
            a.u[v] = *reinterpret_cast<const unsigned*>(actLDS + m16 * lda + k);
        }
        const unsigned* bp =
            reinterpret_cast<const unsigned*>(packB + (((size_t)(chunk0 + kc)) * 32 + lane) * 16);
#pragma unroll
        for (int v = 0; v < 8; ++v) bf.u[v] = bp[v];
        acc = __builtin_amdgcn_wmma_f32_16x16x32_f16(false, a.v, false, bf.v,
                                                     (short)0, acc, false, false);
    }
    return acc;
}

// D layout: lane L holds column n=L&15; vgpr v -> row m = v + 8*(L>>4).
__device__ __forceinline__ void store_leaky_lds(v8f acc, _Float16* dst, int ldd, int nc, int lane)
{
    const int n = lane & 15;
    const int g = lane >> 4;
#pragma unroll
    for (int v = 0; v < 8; ++v)
        dst[(v + 8 * g) * ldd + nc * 16 + n] = (_Float16)leakyf(acc[v]);
}

// ------------------------- weight packing ----------------------------------
// dst layout: chunk = nc*nKc + kc; lane L (n = nc*16 + (L&15), g = L>>4)
// owns 16 halves: K = kc*32 + 16g + h. Zero pad beyond K / Nfull.
__global__ void pack_b_kernel(const float* __restrict__ W, _Float16* __restrict__ dst,
                              int K, int Nfull, int nKc, int nNc)
{
    int gid   = blockIdx.x * blockDim.x + threadIdx.x;
    int total = nNc * nKc * 32;
    if (gid >= total) return;
    int lane  = gid & 31;
    int chunk = gid >> 5;
    int kc = chunk % nKc;
    int nc = chunk / nKc;
    int n  = nc * 16 + (lane & 15);
    int g  = lane >> 4;
    _Float16* d = dst + (size_t)gid * 16;
#pragma unroll
    for (int h = 0; h < 16; ++h) {
        int   k = kc * 32 + g * 16 + h;
        float v = (k < K && n < Nfull) ? W[(size_t)k * Nfull + n] : 0.0f;
        d[h] = (_Float16)v;
    }
}

// ------------------------------ FPS ----------------------------------------
// One block per batch. Running min-distance kept in registers (32 per thread),
// block-wide first-max argmax via LDS tree. Emits selected coordinates.
__global__ __launch_bounds__(256) void fps_kernel(const float* __restrict__ pts, int n,
                                                  int stride, int npoint,
                                                  float* __restrict__ out_xyz)
{
    __shared__ float sVal[256];
    __shared__ int   sIdx[256];
    __shared__ float sCent[3];
    __shared__ int   sFar;
    const int t = threadIdx.x;
    const int b = blockIdx.x;
    const float* P = pts + (size_t)b * n * stride;
    float dloc[32];
#pragma unroll
    for (int j = 0; j < 32; ++j) dloc[j] = 1e10f;
    if (t == 0) sFar = 0;
    __syncthreads();
    for (int it = 0; it < npoint; ++it) {
        const int far = sFar;
        if (t < 3) {
            float c = P[(size_t)far * stride + t];
            out_xyz[((size_t)b * npoint + it) * 3 + t] = c;
            sCent[t] = c;
        }
        __syncthreads();
        const float cx = sCent[0], cy = sCent[1], cz = sCent[2];
        float bv = -1.0f;
        int   bi = 0;
#pragma unroll
        for (int j = 0; j < 32; ++j) {
            int idx = j * 256 + t;
            if (idx < n) {
                float dx = P[(size_t)idx * stride + 0] - cx;
                float dy = P[(size_t)idx * stride + 1] - cy;
                float dz = P[(size_t)idx * stride + 2] - cz;
                float d  = dx * dx + dy * dy + dz * dz;
                float m  = fminf(dloc[j], d);
                dloc[j]  = m;
                if (m > bv) { bv = m; bi = idx; }
            }
        }
        sVal[t] = bv;
        sIdx[t] = bi;
        __syncthreads();
        for (int s = 128; s > 0; s >>= 1) {
            if (t < s) {
                float v2 = sVal[t + s];
                int   i2 = sIdx[t + s];
                if (v2 > sVal[t] || (v2 == sVal[t] && i2 < sIdx[t])) { sVal[t] = v2; sIdx[t] = i2; }
            }
            __syncthreads();
        }
        if (t == 0) sFar = sIdx[0];
        __syncthreads();
    }
}

// ------------------------------ SA1 ----------------------------------------
__global__ __launch_bounds__(64) void sa1_kernel(const float* __restrict__ x,
                                                 const float* __restrict__ new_xyz,
                                                 const float* __restrict__ W1, const float* __restrict__ bb1,
                                                 const float* __restrict__ W2, const float* __restrict__ bb2,
                                                 const float* __restrict__ W3, const float* __restrict__ bb3,
                                                 float* __restrict__ l1_pts)
{
    __shared__ float sNew[3];
    __shared__ int   sIdx[NS1];
    __shared__ float sIn[9];
    __shared__ float sH1[32];
    __shared__ float sH2[32];
    const int t  = threadIdx.x;
    const int bm = blockIdx.x;
    const int b  = bm / NP1;
    if (t < 3) sNew[t] = new_xyz[(size_t)bm * 3 + t];
    __syncthreads();
    if (t == 0) {
        const float nx = sNew[0], ny = sNew[1], nz = sNew[2];
        const float* X = x + (size_t)b * N_ * 9;
        int cnt = 0;
        for (int n = 0; n < N_ && cnt < NS1; ++n) {
            float dx = X[(size_t)n * 9 + 0] - nx;
            float dy = X[(size_t)n * 9 + 1] - ny;
            float dz = X[(size_t)n * 9 + 2] - nz;
            if (dx * dx + dy * dy + dz * dz < R1SQ) sIdx[cnt++] = n;
        }
        if (cnt == 0) { for (int s = 0; s < NS1; ++s) sIdx[s] = N_ - 1; }
        else          { for (int s = cnt; s < NS1; ++s) sIdx[s] = sIdx[0]; }
    }
    __syncthreads();
    float omax = -1e30f;
    for (int s = 0; s < NS1; ++s) {
        if (t < 9) {
            const float* X = x + ((size_t)b * N_ + sIdx[s]) * 9;
            sIn[t] = (t < 3) ? (X[t] - sNew[t]) : X[t];
        }
        __syncthreads();
        if (t < 32) {
            float a = bb1[t];
            for (int i = 0; i < 9; ++i) a += sIn[i] * W1[i * 32 + t];
            sH1[t] = leakyf(a);
        }
        __syncthreads();
        if (t < 32) {
            float a = bb2[t];
            for (int i = 0; i < 32; ++i) a += sH1[i] * W2[i * 32 + t];
            sH2[t] = leakyf(a);
        }
        __syncthreads();
        {
            float a = bb3[t];
            for (int i = 0; i < 32; ++i) a += sH2[i] * W3[i * 64 + t];
            omax = fmaxf(omax, leakyf(a));
        }
        __syncthreads();
    }
    l1_pts[(size_t)bm * 64 + t] = omax;
}

// ------------------------------ SA2 ----------------------------------------
__global__ __launch_bounds__(128) void sa2_kernel(const float* __restrict__ l1_xyz,
                                                  const float* __restrict__ l1_pts,
                                                  const float* __restrict__ new_xyz,
                                                  const float* __restrict__ W1, const float* __restrict__ bb1,
                                                  const float* __restrict__ W2, const float* __restrict__ bb2,
                                                  const float* __restrict__ W3, const float* __restrict__ bb3,
                                                  float* __restrict__ l2_pts)
{
    __shared__ float sNew[3];
    __shared__ int   sIdx[NS2];
    __shared__ float sIn[67];
    __shared__ float sH1[64];
    __shared__ float sH2[64];
    const int t  = threadIdx.x;
    const int bm = blockIdx.x;
    const int b  = bm / NP2;
    if (t < 3) sNew[t] = new_xyz[(size_t)bm * 3 + t];
    __syncthreads();
    if (t == 0) {
        const float nx = sNew[0], ny = sNew[1], nz = sNew[2];
        const float* P = l1_xyz + (size_t)b * NP1 * 3;
        int cnt = 0;
        for (int n2 = 0; n2 < NP1 && cnt < NS2; ++n2) {
            float dx = P[n2 * 3 + 0] - nx;
            float dy = P[n2 * 3 + 1] - ny;
            float dz = P[n2 * 3 + 2] - nz;
            if (dx * dx + dy * dy + dz * dz < R2SQ) sIdx[cnt++] = n2;
        }
        if (cnt == 0) { for (int s = 0; s < NS2; ++s) sIdx[s] = NP1 - 1; }
        else          { for (int s = cnt; s < NS2; ++s) sIdx[s] = sIdx[0]; }
    }
    __syncthreads();
    float omax = -1e30f;
    for (int s = 0; s < NS2; ++s) {
        if (t < 67) {
            int idx = sIdx[s];
            if (t < 3) sIn[t] = l1_xyz[((size_t)b * NP1 + idx) * 3 + t] - sNew[t];
            else       sIn[t] = l1_pts[((size_t)b * NP1 + idx) * 64 + (t - 3)];
        }
        __syncthreads();
        if (t < 64) {
            float a = bb1[t];
            for (int i = 0; i < 67; ++i) a += sIn[i] * W1[i * 64 + t];
            sH1[t] = leakyf(a);
        }
        __syncthreads();
        if (t < 64) {
            float a = bb2[t];
            for (int i = 0; i < 64; ++i) a += sH1[i] * W2[i * 64 + t];
            sH2[t] = leakyf(a);
        }
        __syncthreads();
        {
            float a = bb3[t];
            for (int i = 0; i < 64; ++i) a += sH2[i] * W3[i * 128 + t];
            omax = fmaxf(omax, leakyf(a));
        }
        __syncthreads();
    }
    l2_pts[(size_t)bm * 128 + t] = omax;
}

// ------------------------------ FP3 (WMMA) ---------------------------------
// One wave per 16 l1-points: 3-NN interp from l2 (14 pts), concat l1_pts ->
// 192-dim, MLP 192->256->128 via WMMA.
__global__ __launch_bounds__(32) void fp3_kernel(const float* __restrict__ l1_xyz,
                                                 const float* __restrict__ l2_xyz,
                                                 const float* __restrict__ l1_pts,
                                                 const float* __restrict__ l2_pts,
                                                 const _Float16* __restrict__ pkA, const float* __restrict__ b1,
                                                 const _Float16* __restrict__ pkB, const float* __restrict__ b2,
                                                 float* __restrict__ fp3_out)
{
    __shared__ __align__(16) _Float16 actA[16 * 192];
    __shared__ __align__(16) _Float16 actH[16 * 256];
    __shared__ int   nnI[16 * 3];
    __shared__ float nnW[16 * 3];
    const int lane = threadIdx.x;
    const int b    = blockIdx.x / 18;
    const int base = (blockIdx.x % 18) * 16;

    if (lane < 16) {
        const int p = base + lane;
        const float* P1 = l1_xyz + ((size_t)b * NP1 + p) * 3;
        const float px = P1[0], py = P1[1], pz = P1[2];
        const float* P2 = l2_xyz + (size_t)b * NP2 * 3;
        float d0 = 1e30f, d1 = 1e30f, d2 = 1e30f;
        int   i0 = 0, i1 = 0, i2 = 0;
        for (int j = 0; j < NP2; ++j) {
            float dx = px - P2[j * 3 + 0], dy = py - P2[j * 3 + 1], dz = pz - P2[j * 3 + 2];
            float d = dx * dx + dy * dy + dz * dz;
            if (d < d0)      { d2 = d1; i2 = i1; d1 = d0; i1 = i0; d0 = d; i0 = j; }
            else if (d < d1) { d2 = d1; i2 = i1; d1 = d;  i1 = j; }
            else if (d < d2) { d2 = d;  i2 = j; }
        }
        float w0 = 1.0f / fmaxf(d0, 1e-10f);
        float w1 = 1.0f / fmaxf(d1, 1e-10f);
        float w2 = 1.0f / fmaxf(d2, 1e-10f);
        float s  = 1.0f / (w0 + w1 + w2);
        nnI[lane * 3 + 0] = i0; nnI[lane * 3 + 1] = i1; nnI[lane * 3 + 2] = i2;
        nnW[lane * 3 + 0] = w0 * s; nnW[lane * 3 + 1] = w1 * s; nnW[lane * 3 + 2] = w2 * s;
    }
    __syncthreads();
    for (int e = lane; e < 16 * 192; e += 32) {
        int m = e / 192, c = e - m * 192;
        float val;
        if (c < 128) {
            const float* P = l2_pts + (size_t)b * NP2 * 128;
            val = nnW[m * 3 + 0] * P[(size_t)nnI[m * 3 + 0] * 128 + c]
                + nnW[m * 3 + 1] * P[(size_t)nnI[m * 3 + 1] * 128 + c]
                + nnW[m * 3 + 2] * P[(size_t)nnI[m * 3 + 2] * 128 + c];
        } else {
            val = l1_pts[((size_t)b * NP1 + base + m) * 64 + (c - 128)];
        }
        actA[m * 192 + c] = (_Float16)val;
    }
    __syncthreads();
    for (int nc = 0; nc < 16; ++nc) {
        v8f acc = wave_gemm_tile(actA, 192, pkA, nc * 6, 6, b1[nc * 16 + (lane & 15)], lane);
        store_leaky_lds(acc, actH, 256, nc, lane);
    }
    __syncthreads();
    const int n = lane & 15, g = lane >> 4;
    for (int nc = 0; nc < 8; ++nc) {
        v8f acc = wave_gemm_tile(actH, 256, pkB, nc * 8, 8, b2[nc * 16 + n], lane);
#pragma unroll
        for (int v = 0; v < 8; ++v) {
            int m = v + 8 * g;
            fp3_out[((size_t)b * NP1 + base + m) * 128 + nc * 16 + n] = leakyf(acc[v]);
        }
    }
}

// ---------------------- fused FP4 + head (WMMA) ----------------------------
// One wave per 16 l0-points: 3-NN over 288 l1 pts, interp 128 + 6 raw feats
// -> 134 (pad 160), MLP 134->128->128->128, head 128->128 -> 13. All
// intermediates stay in LDS; only the 13-class logits hit HBM.
__global__ __launch_bounds__(32) void fp4_head_kernel(const float* __restrict__ x,
                                                      const float* __restrict__ l1_xyz,
                                                      const float* __restrict__ fp3o,
                                                      const _Float16* __restrict__ pkW1, const float* __restrict__ b1,
                                                      const _Float16* __restrict__ pkW2, const float* __restrict__ b2,
                                                      const _Float16* __restrict__ pkW3, const float* __restrict__ b3,
                                                      const _Float16* __restrict__ pkH1, const float* __restrict__ bh1,
                                                      const _Float16* __restrict__ pkH2, const float* __restrict__ bh2,
                                                      float* __restrict__ out)
{
    __shared__ __align__(16) _Float16 actA[16 * 160];
    __shared__ __align__(16) _Float16 buf0[16 * 128];
    __shared__ __align__(16) _Float16 buf1[16 * 128];
    __shared__ int   nnI[16 * 3];
    __shared__ float nnW[16 * 3];
    const int lane = threadIdx.x;
    const int b    = blockIdx.x >> 9;          // 512 tiles of 16 per batch
    const int base = (blockIdx.x & 511) * 16;

    if (lane < 16) {
        const float* xp = x + ((size_t)b * N_ + base + lane) * 9;
        const float px = xp[0], py = xp[1], pz = xp[2];
        const float* L1 = l1_xyz + (size_t)b * NP1 * 3;
        float d0 = 1e30f, d1 = 1e30f, d2 = 1e30f;
        int   i0 = 0, i1 = 0, i2 = 0;
        for (int j = 0; j < NP1; ++j) {
            float dx = px - L1[j * 3 + 0], dy = py - L1[j * 3 + 1], dz = pz - L1[j * 3 + 2];
            float d = dx * dx + dy * dy + dz * dz;
            if (d < d0)      { d2 = d1; i2 = i1; d1 = d0; i1 = i0; d0 = d; i0 = j; }
            else if (d < d1) { d2 = d1; i2 = i1; d1 = d;  i1 = j; }
            else if (d < d2) { d2 = d;  i2 = j; }
        }
        float w0 = 1.0f / fmaxf(d0, 1e-10f);
        float w1 = 1.0f / fmaxf(d1, 1e-10f);
        float w2 = 1.0f / fmaxf(d2, 1e-10f);
        float s  = 1.0f / (w0 + w1 + w2);
        nnI[lane * 3 + 0] = i0; nnI[lane * 3 + 1] = i1; nnI[lane * 3 + 2] = i2;
        nnW[lane * 3 + 0] = w0 * s; nnW[lane * 3 + 1] = w1 * s; nnW[lane * 3 + 2] = w2 * s;
    }
    __syncthreads();
    for (int e = lane; e < 16 * 160; e += 32) {
        int m = e / 160, c = e - m * 160;
        float val;
        if (c < 128) {
            const float* P = fp3o + (size_t)b * NP1 * 128;
            val = nnW[m * 3 + 0] * P[(size_t)nnI[m * 3 + 0] * 128 + c]
                + nnW[m * 3 + 1] * P[(size_t)nnI[m * 3 + 1] * 128 + c]
                + nnW[m * 3 + 2] * P[(size_t)nnI[m * 3 + 2] * 128 + c];
        } else if (c < 134) {
            val = x[((size_t)b * N_ + base + m) * 9 + 3 + (c - 128)];
        } else {
            val = 0.0f;
        }
        actA[m * 160 + c] = (_Float16)val;
    }
    __syncthreads();
    const int n = lane & 15, g = lane >> 4;
    for (int nc = 0; nc < 8; ++nc) {                       // 134(pad160) -> 128
        v8f acc = wave_gemm_tile(actA, 160, pkW1, nc * 5, 5, b1[nc * 16 + n], lane);
        store_leaky_lds(acc, buf0, 128, nc, lane);
    }
    __syncthreads();
    for (int nc = 0; nc < 8; ++nc) {                       // 128 -> 128
        v8f acc = wave_gemm_tile(buf0, 128, pkW2, nc * 4, 4, b2[nc * 16 + n], lane);
        store_leaky_lds(acc, buf1, 128, nc, lane);
    }
    __syncthreads();
    for (int nc = 0; nc < 8; ++nc) {                       // 128 -> 128
        v8f acc = wave_gemm_tile(buf1, 128, pkW3, nc * 4, 4, b3[nc * 16 + n], lane);
        store_leaky_lds(acc, buf0, 128, nc, lane);
    }
    __syncthreads();
    for (int nc = 0; nc < 8; ++nc) {                       // head1 128 -> 128
        v8f acc = wave_gemm_tile(buf0, 128, pkH1, nc * 4, 4, bh1[nc * 16 + n], lane);
        store_leaky_lds(acc, buf1, 128, nc, lane);
    }
    __syncthreads();
    {                                                      // head2 128 -> 13 (no act)
        float bv = (n < NCLS) ? bh2[n] : 0.0f;
        v8f acc  = wave_gemm_tile(buf1, 128, pkH2, 0, 4, bv, lane);
        if (n < NCLS) {
#pragma unroll
            for (int v = 0; v < 8; ++v) {
                int m = v + 8 * g;
                out[((size_t)b * N_ + base + m) * NCLS + n] = acc[v];
            }
        }
    }
}

// ------------------------------ host ---------------------------------------
extern "C" void kernel_launch(void* const* d_in, const int* in_sizes, int n_in,
                              void* d_out, int out_size, void* d_ws, size_t ws_size,
                              hipStream_t stream)
{
    (void)in_sizes; (void)n_in; (void)out_size; (void)ws_size;
    const float* x    = (const float*)d_in[0];
    const float* s1W1 = (const float*)d_in[1],  *s1b1 = (const float*)d_in[2];
    const float* s1W2 = (const float*)d_in[3],  *s1b2 = (const float*)d_in[4];
    const float* s1W3 = (const float*)d_in[5],  *s1b3 = (const float*)d_in[6];
    const float* s2W1 = (const float*)d_in[7],  *s2b1 = (const float*)d_in[8];
    const float* s2W2 = (const float*)d_in[9],  *s2b2 = (const float*)d_in[10];
    const float* s2W3 = (const float*)d_in[11], *s2b3 = (const float*)d_in[12];
    const float* f3W1 = (const float*)d_in[13], *f3b1 = (const float*)d_in[14];
    const float* f3W2 = (const float*)d_in[15], *f3b2 = (const float*)d_in[16];
    const float* f4W1 = (const float*)d_in[17], *f4b1 = (const float*)d_in[18];
    const float* f4W2 = (const float*)d_in[19], *f4b2 = (const float*)d_in[20];
    const float* f4W3 = (const float*)d_in[21], *f4b3 = (const float*)d_in[22];
    const float* h1W  = (const float*)d_in[23], *h1b  = (const float*)d_in[24];
    const float* h2W  = (const float*)d_in[25], *h2b  = (const float*)d_in[26];
    float* out = (float*)d_out;

    char*  ws = (char*)d_ws;
    size_t o  = 0;
    auto take = [&](size_t bytes) -> char* {
        char* r = ws + o;
        o = (o + bytes + 255) & ~(size_t)255;
        return r;
    };
    float* l1_xyz = (float*)take((size_t)B_ * NP1 * 3 * 4);
    float* l1_pts = (float*)take((size_t)B_ * NP1 * 64 * 4);
    float* l2_xyz = (float*)take((size_t)B_ * NP2 * 3 * 4);
    float* l2_pts = (float*)take((size_t)B_ * NP2 * 128 * 4);
    float* fp3o   = (float*)take((size_t)B_ * NP1 * 128 * 4);
    _Float16* pk_f3a = (_Float16*)take((size_t)16 * 6 * 32 * 16 * 2);
    _Float16* pk_f3b = (_Float16*)take((size_t)8  * 8 * 32 * 16 * 2);
    _Float16* pk_f4a = (_Float16*)take((size_t)8  * 5 * 32 * 16 * 2);
    _Float16* pk_f4b = (_Float16*)take((size_t)8  * 4 * 32 * 16 * 2);
    _Float16* pk_f4c = (_Float16*)take((size_t)8  * 4 * 32 * 16 * 2);
    _Float16* pk_h1  = (_Float16*)take((size_t)8  * 4 * 32 * 16 * 2);
    _Float16* pk_h2  = (_Float16*)take((size_t)1  * 4 * 32 * 16 * 2);

    auto packLaunch = [&](const float* W, _Float16* dst, int K, int Nfull, int nKc, int nNc) {
        int total = nNc * nKc * 32;
        pack_b_kernel<<<(total + 63) / 64, 64, 0, stream>>>(W, dst, K, Nfull, nKc, nNc);
    };
    packLaunch(f3W1, pk_f3a, 192, 256, 6, 16);
    packLaunch(f3W2, pk_f3b, 256, 128, 8, 8);
    packLaunch(f4W1, pk_f4a, 134, 128, 5, 8);
    packLaunch(f4W2, pk_f4b, 128, 128, 4, 8);
    packLaunch(f4W3, pk_f4c, 128, 128, 4, 8);
    packLaunch(h1W,  pk_h1,  128, 128, 4, 8);
    packLaunch(h2W,  pk_h2,  128, NCLS, 4, 1);

    fps_kernel<<<B_, 256, 0, stream>>>(x, N_, 9, NP1, l1_xyz);
    sa1_kernel<<<B_ * NP1, 64, 0, stream>>>(x, l1_xyz, s1W1, s1b1, s1W2, s1b2, s1W3, s1b3, l1_pts);
    fps_kernel<<<B_, 256, 0, stream>>>(l1_xyz, NP1, 3, NP2, l2_xyz);
    sa2_kernel<<<B_ * NP2, 128, 0, stream>>>(l1_xyz, l1_pts, l2_xyz,
                                             s2W1, s2b1, s2W2, s2b2, s2W3, s2b3, l2_pts);
    fp3_kernel<<<B_ * (NP1 / 16), 32, 0, stream>>>(l1_xyz, l2_xyz, l1_pts, l2_pts,
                                                   pk_f3a, f3b1, pk_f3b, f3b2, fp3o);
    fp4_head_kernel<<<B_ * (N_ / 16), 32, 0, stream>>>(x, l1_xyz, fp3o,
                                                       pk_f4a, f4b1, pk_f4b, f4b2, pk_f4c, f4b3,
                                                       pk_h1, h1b, pk_h2, h2b, out);
}
